// MultiHeadAttention_6992206758458
// MI455X (gfx1250) — matrix operand order repaired and checked by reading, not compiled, not run on hardware
//
#include <hip/hip_runtime.h>
#include <hip/hip_bf16.h>

typedef __attribute__((ext_vector_type(16))) __bf16 v16bf;
typedef __attribute__((ext_vector_type(8)))  __bf16 v8bf;
typedef __attribute__((ext_vector_type(8)))  float  v8f;

static constexpr int B_  = 2;
static constexpr int S_  = 2048;
static constexpr int H_  = 1024;
static constexpr int NH  = 16;
static constexpr int HD  = 64;
static constexpr int M_  = B_ * S_;          // 4096 rows
static constexpr float SCALE = 0.125f;       // 1/sqrt(64)

// ---------- helpers ----------

__device__ __forceinline__ __bf16 f2bf(float f) {
  union { float f; unsigned u; } c{f};
  unsigned r = c.u + 0x7FFFu + ((c.u >> 16) & 1u);   // round-to-nearest-even
  unsigned short h = (unsigned short)(r >> 16);
  return __builtin_bit_cast(__bf16, h);
}

__device__ __forceinline__ v8f zero8() {
  v8f z;
#pragma unroll
  for (int i = 0; i < 8; ++i) z[i] = 0.0f;
  return z;
}

__device__ __forceinline__ v8f wmma_bf16(v16bf a, v16bf b, v8f c) {
  return __builtin_amdgcn_wmma_f32_16x16x32_bf16(false, a, false, b, (short)0, c,
                                                 false, false);
}

// Async global->LDS copy, 16 bytes per lane, tracked by ASYNCcnt.
// LDS dest is the wave-relative byte offset (low 32 bits of the generic pointer).
__device__ __forceinline__ void async_copy_b128(const void* gptr, void* lds) {
  unsigned loff = (unsigned)(size_t)lds;
  unsigned long long g = (unsigned long long)(size_t)gptr;
  asm volatile("global_load_async_to_lds_b128 %0, %1, off"
               :: "v"(loff), "v"(g) : "memory");
}

__device__ __forceinline__ void wait_async0() {
  asm volatile("s_wait_asynccnt 0x0" ::: "memory");
}

// ====== f32 -> bf16 bulk conversion (for weights) ======
__global__ __launch_bounds__(256) void cvt_f32_bf16_kernel(
    const float* __restrict__ in, __bf16* __restrict__ out, int n) {
  int i = (blockIdx.x * 256 + threadIdx.x) * 8;
  if (i + 8 <= n) {
    v8f x = *(const v8f*)(in + i);
    v8bf h;
#pragma unroll
    for (int j = 0; j < 8; ++j) h[j] = f2bf(x[j]);
    *(v8bf*)(out + i) = h;
  }
}

// ================= GEMM: Out[M,N] = A[M,K] * W[N,K]^T + bias[N] =================
// Block: 256 threads (8 waves). Block tile 128(M) x 64(N), K step 32,
// double-buffered LDS with async global->LDS staging for bf16 operands.
// Wave (wm 0..3, wn 0..1) computes a 32x32 output: 4 WMMA accumulators.
template <bool A_BF16, bool HEAD_OUT>
__global__ __launch_bounds__(256) void gemm_bias_kernel(
    const void* __restrict__ Aptr, const __bf16* __restrict__ Wb,
    const float* __restrict__ bias, void* __restrict__ Out,
    int M, int N, int K) {
  __shared__ alignas(64) __bf16 As[2][128][32];   // 16 KB
  __shared__ alignas(64) __bf16 Bs[2][64][32];    //  8 KB

  const int t    = threadIdx.x;
  const int w    = t >> 5;
  const int lane = t & 31;
  const int lm   = lane & 15;
  const int wm   = w & 3;
  const int wn   = w >> 2;
  const int m0   = blockIdx.y * 128;
  const int n0   = blockIdx.x * 64;

  // staging coords: A: 128 rows x 32 cols, 16 elems/thread
  const int arow = t >> 1;
  const int acol = (t & 1) * 16;
  // B: 64 rows x 32 cols, threads 0..127
  const int brow = t >> 1;
  const int bcol = (t & 1) * 16;

  // contiguous k order (k = 16h + s) lane-half offsets, same for A and B
  const int akb = (lane < 16) ? 0 : 16;
  const int bkb = (lane < 16) ? 0 : 16;

  auto stage = [&](int buf, int k0) {
    if constexpr (A_BF16) {
      const __bf16* Ag = (const __bf16*)Aptr + (size_t)(m0 + arow) * K + k0 + acol;
      async_copy_b128(Ag,     &As[buf][arow][acol]);
      async_copy_b128(Ag + 8, &As[buf][arow][acol + 8]);
    } else {
      const float* Ag = (const float*)Aptr + (size_t)(m0 + arow) * K + k0 + acol;
#pragma unroll
      for (int i = 0; i < 16; i += 8) {
        v8bf h;
#pragma unroll
        for (int j = 0; j < 8; ++j) h[j] = f2bf(Ag[i + j]);
        *(v8bf*)&As[buf][arow][acol + i] = h;
      }
    }
    if (t < 128) {
      const __bf16* Wg = Wb + (size_t)(n0 + brow) * K + k0 + bcol;
      async_copy_b128(Wg,     &Bs[buf][brow][bcol]);
      async_copy_b128(Wg + 8, &Bs[buf][brow][bcol + 8]);
    }
  };

  v8f acc[2][2];
#pragma unroll
  for (int i = 0; i < 2; ++i)
#pragma unroll
    for (int j = 0; j < 2; ++j) acc[i][j] = zero8();

  const int steps = K >> 5;
  stage(0, 0);

  for (int it = 0; it < steps; ++it) {
    wait_async0();
    __syncthreads();
    if (it + 1 < steps) stage((it + 1) & 1, (it + 1) * 32);

    const int buf = it & 1;
    v16bf a0, a1;
    {
      v8bf lo = *(const v8bf*)&As[buf][wm * 32 + lm][akb];
      v8bf hi = *(const v8bf*)&As[buf][wm * 32 + lm][akb + 8];
      v8bf lo1 = *(const v8bf*)&As[buf][wm * 32 + 16 + lm][akb];
      v8bf hi1 = *(const v8bf*)&As[buf][wm * 32 + 16 + lm][akb + 8];
#pragma unroll
      for (int i = 0; i < 8; ++i) {
        a0[i] = lo[i];  a0[8 + i] = hi[i];
        a1[i] = lo1[i]; a1[8 + i] = hi1[i];
      }
    }
    v16bf b0 = *(const v16bf*)&Bs[buf][wn * 32 + lm][bkb];
    v16bf b1 = *(const v16bf*)&Bs[buf][wn * 32 + 16 + lm][bkb];

    acc[0][0] = wmma_bf16(a0, b0, acc[0][0]);
    acc[0][1] = wmma_bf16(a0, b1, acc[0][1]);
    acc[1][0] = wmma_bf16(a1, b0, acc[1][0]);
    acc[1][1] = wmma_bf16(a1, b1, acc[1][1]);
  }

  // Epilogue. C/D layout: VGPR r, lanes 0-15 -> row r, lanes 16-31 -> row r+8.
  const int rowg = (lane >> 4) * 8;
#pragma unroll
  for (int at = 0; at < 2; ++at) {
#pragma unroll
    for (int nt = 0; nt < 2; ++nt) {
      v8f c = acc[at][nt];
      const int n  = n0 + wn * 32 + nt * 16 + lm;
      const float bv = bias[n];
#pragma unroll
      for (int r = 0; r < 8; ++r) {
        const int m = m0 + wm * 32 + at * 16 + rowg + r;
        const float val = c[r] + bv;
        if constexpr (HEAD_OUT) {
          const int bb = m >> 11;          // / S_
          const int s  = m & (S_ - 1);
          const int hh = n >> 6;           // / HD
          const int d  = n & (HD - 1);
          ((__bf16*)Out)[(((size_t)(bb * NH + hh)) * S_ + s) * HD + d] = f2bf(val);
        } else {
          ((float*)Out)[(size_t)m * N + n] = val;
        }
      }
    }
  }
}

// ================= Flash attention per (b,h) =================
// Block: 256 threads = 8 waves; each wave owns a 16-row Q tile (128 q rows/block).
// KV double-buffered through LDS (K via async copies), online softmax,
// P re-laid out C->A through per-wave LDS with a DScnt drain.
__global__ __launch_bounds__(256) void flash_attn_kernel(
    const __bf16* __restrict__ Qh, const __bf16* __restrict__ Kh,
    const __bf16* __restrict__ Vh, const int* __restrict__ mask,
    __bf16* __restrict__ Y) {
  __shared__ alignas(64) __bf16 Ks[2][64][64];     // 16 KB
  __shared__ alignas(64) __bf16 Vt[2][64][64];     // 16 KB, [d][kv]
  __shared__ alignas(64) __bf16 Ps[8][16][32];     //  8 KB

  const int t    = threadIdx.x;
  const int w    = t >> 5;
  const int lane = t & 31;
  const int lm   = lane & 15;
  const int bh   = blockIdx.y;
  const int b    = bh >> 4;
  const int qr   = blockIdx.x * 128 + w * 16;

  const __bf16* qbase = Qh + ((size_t)bh * S_ + qr) * HD;
  const __bf16* kbase = Kh + (size_t)bh * S_ * HD;
  const __bf16* vbase = Vh + (size_t)bh * S_ * HD;
  const int*    mrow  = mask + b * S_;

  const int akb = (lane < 16) ? 0 : 16;
  const int bkb = (lane < 16) ? 0 : 16;

  auto stageKV = [&](int buf, int j0) {
    const int jr = t >> 2;
    const int c  = (t & 3) * 16;
    const __bf16* kg = kbase + (size_t)(j0 + jr) * HD + c;
    async_copy_b128(kg,     &Ks[buf][jr][c]);
    async_copy_b128(kg + 8, &Ks[buf][jr][c + 8]);
    const __bf16* vg = vbase + (size_t)(j0 + jr) * HD + c;
    v8bf v0 = *(const v8bf*)vg;
    v8bf v1 = *(const v8bf*)(vg + 8);
#pragma unroll
    for (int i = 0; i < 8; ++i) {
      Vt[buf][c + i][jr]     = v0[i];
      Vt[buf][c + 8 + i][jr] = v1[i];
    }
  };

  // Q fragments for d=0..31 and d=32..63, resident in VGPRs.
  v16bf qa0, qa1;
  {
    const __bf16* qp = qbase + (size_t)lm * HD;
    v8bf x0 = *(const v8bf*)(qp + akb);
    v8bf x1 = *(const v8bf*)(qp + akb + 8);
    v8bf x2 = *(const v8bf*)(qp + 32 + akb);
    v8bf x3 = *(const v8bf*)(qp + 32 + akb + 8);
#pragma unroll
    for (int i = 0; i < 8; ++i) {
      qa0[i] = x0[i]; qa0[8 + i] = x1[i];
      qa1[i] = x2[i]; qa1[8 + i] = x3[i];
    }
  }

  v8f O[4];
#pragma unroll
  for (int dt = 0; dt < 4; ++dt) O[dt] = zero8();
  float mi[8], li[8];
#pragma unroll
  for (int r = 0; r < 8; ++r) { mi[r] = -1e30f; li[r] = 0.0f; }

  constexpr int KVBLKS = S_ / 64;
  stageKV(0, 0);

  for (int jb = 0; jb < KVBLKS; ++jb) {
    wait_async0();
    __syncthreads();
    if (jb + 1 < KVBLKS) stageKV((jb + 1) & 1, (jb + 1) * 64);

    const int buf = jb & 1;
    const int j0  = jb * 64;

#pragma unroll
    for (int pair = 0; pair < 2; ++pair) {
      const int c0 = pair * 32;

      // ---- scores: two 16x16 tiles ----
      v8f se = zero8(), so = zero8();
      {
        const int je = c0 + lm, jo = c0 + 16 + lm;
        v16bf ke0 = *(const v16bf*)&Ks[buf][je][bkb];
        v16bf ke1 = *(const v16bf*)&Ks[buf][je][32 + bkb];
        v16bf ko0 = *(const v16bf*)&Ks[buf][jo][bkb];
        v16bf ko1 = *(const v16bf*)&Ks[buf][jo][32 + bkb];
        se = wmma_bf16(qa0, ke0, se);
        se = wmma_bf16(qa1, ke1, se);
        so = wmma_bf16(qa0, ko0, so);
        so = wmma_bf16(qa1, ko1, so);
      }

      const bool oke = mrow[j0 + c0 + lm] != 0;
      const bool oko = mrow[j0 + c0 + 16 + lm] != 0;
#pragma unroll
      for (int r = 0; r < 8; ++r) {
        se[r] = oke ? se[r] * SCALE : -1e9f;
        so[r] = oko ? so[r] * SCALE : -1e9f;
      }

      // ---- online softmax over these 32 columns ----
#pragma unroll
      for (int r = 0; r < 8; ++r) {
        float mx = fmaxf(se[r], so[r]);
#pragma unroll
        for (int off = 1; off < 16; off <<= 1)
          mx = fmaxf(mx, __shfl_xor(mx, off, 32));
        const float mnew  = fmaxf(mi[r], mx);
        const float alpha = __expf(mi[r] - mnew);
        mi[r] = mnew;
        const float pe = __expf(se[r] - mnew);
        const float po = __expf(so[r] - mnew);
        se[r] = pe; so[r] = po;
        float rs = pe + po;
#pragma unroll
        for (int off = 1; off < 16; off <<= 1)
          rs += __shfl_xor(rs, off, 32);
        li[r] = li[r] * alpha + rs;
#pragma unroll
        for (int dt = 0; dt < 4; ++dt) O[dt][r] *= alpha;
      }

      // ---- C-layout -> A-layout for P via per-wave LDS tile ----
      {
        const int rowg = (lane >> 4) * 8;
#pragma unroll
        for (int r = 0; r < 8; ++r) {
          Ps[w][rowg + r][lm]      = f2bf(se[r]);
          Ps[w][rowg + r][16 + lm] = f2bf(so[r]);
        }
      }
      asm volatile("s_wait_dscnt 0x0" ::: "memory");   // same-wave DS drain

      v16bf pa;
      {
        const __bf16* pp = &Ps[w][lm][0];
        v8bf lo = *(const v8bf*)(pp + akb);
        v8bf hi = *(const v8bf*)(pp + akb + 8);
#pragma unroll
        for (int i = 0; i < 8; ++i) { pa[i] = lo[i]; pa[8 + i] = hi[i]; }
      }

      // ---- O += P(16x32) x V(32x64) ----
#pragma unroll
      for (int dt = 0; dt < 4; ++dt) {
        v16bf vb = *(const v16bf*)&Vt[buf][dt * 16 + lm][c0 + bkb];
        O[dt] = wmma_bf16(pa, vb, O[dt]);
      }
    }
  }

  // ---- normalize and scatter to Y[b, s, h*64+d] (bf16) ----
  const int h    = bh & 15;
  const int rowg = (lane >> 4) * 8;
  __bf16* yout = Y + ((size_t)b * S_ + qr) * H_ + h * HD;
#pragma unroll
  for (int r = 0; r < 8; ++r) {
    const float inv = 1.0f / li[r];
#pragma unroll
    for (int dt = 0; dt < 4; ++dt) {
      yout[(size_t)(rowg + r) * H_ + dt * 16 + lm] = f2bf(O[dt][r] * inv);
    }
  }
}

// ================= launch =================
extern "C" void kernel_launch(void* const* d_in, const int* in_sizes, int n_in,
                              void* d_out, int out_size, void* d_ws, size_t ws_size,
                              hipStream_t stream) {
  const float* q    = (const float*)d_in[0];
  const float* k    = (const float*)d_in[1];
  const float* v    = (const float*)d_in[2];
  const int*   mask = (const int*)d_in[3];
  const float* Wq   = (const float*)d_in[4];
  const float* bq   = (const float*)d_in[5];
  const float* Wk   = (const float*)d_in[6];
  const float* bk   = (const float*)d_in[7];
  const float* Wv   = (const float*)d_in[8];
  const float* bv   = (const float*)d_in[9];
  const float* Wo   = (const float*)d_in[10];
  const float* bo   = (const float*)d_in[11];

  const size_t wElems = (size_t)H_ * H_;             // 1M per weight
  const size_t aElems = (size_t)B_ * NH * S_ * HD;   // 4M per activation

  __bf16* Wqb = (__bf16*)d_ws;
  __bf16* Wkb = Wqb + wElems;
  __bf16* Wvb = Wkb + wElems;
  __bf16* Wob = Wvb + wElems;
  __bf16* qh  = Wob + wElems;
  __bf16* kh  = qh + aElems;
  __bf16* vh  = kh + aElems;
  __bf16* y   = vh + aElems;                         // [B, S, H] bf16

  dim3 blk(256);
  dim3 gCvt((unsigned)(wElems / (256 * 8)));         // 512 blocks
  dim3 gGemm(H_ / 64, M_ / 128);                     // (16, 32)
  dim3 gAttn(S_ / 128, B_ * NH);                     // (16, 32)

  cvt_f32_bf16_kernel<<<gCvt, blk, 0, stream>>>(Wq, Wqb, (int)wElems);
  cvt_f32_bf16_kernel<<<gCvt, blk, 0, stream>>>(Wk, Wkb, (int)wElems);
  cvt_f32_bf16_kernel<<<gCvt, blk, 0, stream>>>(Wv, Wvb, (int)wElems);
  cvt_f32_bf16_kernel<<<gCvt, blk, 0, stream>>>(Wo, Wob, (int)wElems);

  gemm_bias_kernel<false, true><<<gGemm, blk, 0, stream>>>(q, Wqb, bq, qh, M_, H_, H_);
  gemm_bias_kernel<false, true><<<gGemm, blk, 0, stream>>>(k, Wkb, bk, kh, M_, H_, H_);
  gemm_bias_kernel<false, true><<<gGemm, blk, 0, stream>>>(v, Wvb, bv, vh, M_, H_, H_);

  flash_attn_kernel<<<gAttn, blk, 0, stream>>>(qh, kh, vh, mask, y);

  gemm_bias_kernel<true, false><<<gGemm, blk, 0, stream>>>(y, Wob, bo, d_out, M_, H_, H_);
}
